// WaveletKANLinear_17411797418704
// MI455X (gfx1250) — compile-verified, matrix operand order checked by reading, not run on hardware
//
#include <hip/hip_runtime.h>
#include <hip/hip_bf16.h>

#define IN_F  512
#define OUT_F 512
#define BATCH 1024

typedef __attribute__((ext_vector_type(16))) _Float16 v16h;
typedef __attribute__((ext_vector_type(8)))  _Float16 v8h;
typedef __attribute__((ext_vector_type(8)))  float    v8f;

// fast silu: v * rcp(1 + exp(-v)) -- v_exp_f32 + v_rcp_f32 + mul/add only.
__device__ __forceinline__ float silu_f(float v) {
    return v * __builtin_amdgcn_rcpf(1.0f + __expf(-v));
}

// ---------------------------------------------------------------------------
// Elementwise converters into workspace (f16 operands for the WMMA GEMM).
// Each thread handles 8 contiguous elements: 2x b128 loads -> 1x b128 store.
// ---------------------------------------------------------------------------
__global__ __launch_bounds__(256) void silu_cvt_f16(
    const float* __restrict__ in, _Float16* __restrict__ out)
{
    const int i = blockIdx.x * 256 + threadIdx.x;     // 8 elems per thread
    const float4 a = ((const float4*)in)[i * 2];
    const float4 b = ((const float4*)in)[i * 2 + 1];
    v8h h;
    h[0] = (_Float16)silu_f(a.x); h[1] = (_Float16)silu_f(a.y);
    h[2] = (_Float16)silu_f(a.z); h[3] = (_Float16)silu_f(a.w);
    h[4] = (_Float16)silu_f(b.x); h[5] = (_Float16)silu_f(b.y);
    h[6] = (_Float16)silu_f(b.z); h[7] = (_Float16)silu_f(b.w);
    ((v8h*)out)[i] = h;
}

__global__ __launch_bounds__(256) void cvt_f16(
    const float* __restrict__ in, _Float16* __restrict__ out)
{
    const int i = blockIdx.x * 256 + threadIdx.x;
    const float4 a = ((const float4*)in)[i * 2];
    const float4 b = ((const float4*)in)[i * 2 + 1];
    v8h h;
    h[0] = (_Float16)a.x; h[1] = (_Float16)a.y;
    h[2] = (_Float16)a.z; h[3] = (_Float16)a.w;
    h[4] = (_Float16)b.x; h[5] = (_Float16)b.y;
    h[6] = (_Float16)b.z; h[7] = (_Float16)b.w;
    ((v8h*)out)[i] = h;
}

// A/B fragment for v_wmma_f32_16x16x32_f16 (ISA 7.12.2):
//   row = lane & 15, kbase = (lane>>4)*8,
//   halfs 0..7 -> K=kbase+0..7, halfs 8..15 -> K=kbase+16..23
__device__ __forceinline__ v16h load_hfrag(const _Float16* __restrict__ row, int kb) {
    const v8h lo = *(const v8h*)(row + kb);        // one global_load_b128
    const v8h hi = *(const v8h*)(row + kb + 16);   // one global_load_b128
    v16h f;
#pragma unroll
    for (int j = 0; j < 8; ++j) { f[j] = lo[j]; f[8 + j] = hi[j]; }
    return f;
}

// ---------------------------------------------------------------------------
// Fast GEMM: out = hx @ hw^T + bias, f16 operands from workspace.
// 8 waves/block; each wave owns a 64x16 tile: 4 accumulators share 1 B frag
// per K-step. Software-pipelined: fragments for step k+1 are issued before
// the WMMAs consuming step k, so the loads stay in flight under the XDL ops.
// ---------------------------------------------------------------------------
__global__ __launch_bounds__(256) void base_gemm_wmma_h(
    const _Float16* __restrict__ hx,   // [BATCH, IN_F] = silu(x) in f16
    const _Float16* __restrict__ hw,   // [OUT_F, IN_F] in f16
    const float* __restrict__ bias,    // [OUT_F]
    float* __restrict__ out)           // [BATCH, OUT_F]
{
    const int lane = threadIdx.x & 31;
    const int wave = threadIdx.x >> 5;
    const int tile = blockIdx.x * 8 + wave;        // 512 tiles
    const int tile_m = (tile & 15) * 64;           // 16 M-tiles of 64
    const int tile_n = (tile >> 4) * 16;           // 32 N-tiles of 16

    const int r   = lane & 15;
    const int kb0 = (lane >> 4) * 8;

    const _Float16* xr = hx + (size_t)(tile_m + r) * IN_F;
    const _Float16* wr = hw + (size_t)(tile_n + r) * IN_F;

    v8f acc0 = {}, acc1 = {}, acc2 = {}, acc3 = {};

    // Prologue: fragments for k0 = 0.
    v16h bf = load_hfrag(wr, kb0);
    v16h a0 = load_hfrag(xr,             kb0);
    v16h a1 = load_hfrag(xr + 16 * IN_F, kb0);
    v16h a2 = load_hfrag(xr + 32 * IN_F, kb0);
    v16h a3 = load_hfrag(xr + 48 * IN_F, kb0);

    for (int k0 = 0; k0 < IN_F - 32; k0 += 32) {
        const int kb = k0 + 32 + kb0;
        // Issue next-step loads first...
        const v16h nbf = load_hfrag(wr, kb);
        const v16h na0 = load_hfrag(xr,             kb);
        const v16h na1 = load_hfrag(xr + 16 * IN_F, kb);
        const v16h na2 = load_hfrag(xr + 32 * IN_F, kb);
        const v16h na3 = load_hfrag(xr + 48 * IN_F, kb);
        // ...then consume current-step fragments.
        acc0 = __builtin_amdgcn_wmma_f32_16x16x32_f16(false, a0, false, bf, (short)0, acc0, false, false);
        acc1 = __builtin_amdgcn_wmma_f32_16x16x32_f16(false, a1, false, bf, (short)0, acc1, false, false);
        acc2 = __builtin_amdgcn_wmma_f32_16x16x32_f16(false, a2, false, bf, (short)0, acc2, false, false);
        acc3 = __builtin_amdgcn_wmma_f32_16x16x32_f16(false, a3, false, bf, (short)0, acc3, false, false);
        bf = nbf; a0 = na0; a1 = na1; a2 = na2; a3 = na3;
    }
    // Epilogue: last K-step.
    acc0 = __builtin_amdgcn_wmma_f32_16x16x32_f16(false, a0, false, bf, (short)0, acc0, false, false);
    acc1 = __builtin_amdgcn_wmma_f32_16x16x32_f16(false, a1, false, bf, (short)0, acc1, false, false);
    acc2 = __builtin_amdgcn_wmma_f32_16x16x32_f16(false, a2, false, bf, (short)0, acc2, false, false);
    acc3 = __builtin_amdgcn_wmma_f32_16x16x32_f16(false, a3, false, bf, (short)0, acc3, false, false);

    // C/D layout: VGPR rr -> M = rr (lanes 0-15) / rr+8 (lanes 16-31), N = lane&15
    const int n  = lane & 15;
    const int mh = (lane >> 4) * 8;
    const float bv = bias[tile_n + n];
    float* o = out + (size_t)(tile_m + mh) * OUT_F + tile_n + n;
#pragma unroll
    for (int rr = 0; rr < 8; ++rr) {
        o[(size_t)rr * OUT_F]        = acc0[rr] + bv;
        o[(size_t)(rr + 16) * OUT_F] = acc1[rr] + bv;
        o[(size_t)(rr + 32) * OUT_F] = acc2[rr] + bv;
        o[(size_t)(rr + 48) * OUT_F] = acc3[rr] + bv;
    }
}

// ---------------------------------------------------------------------------
// Fallback fused GEMM (used only if workspace is too small): converts fp32 ->
// silu -> f16 inline. Cheap-silu version (rcp, not IEEE divide).
// ---------------------------------------------------------------------------
__device__ __forceinline__ v16h make_frag_f32(const float* __restrict__ row, int kb, bool do_silu) {
    const float4 p0 = *(const float4*)(row + kb);
    const float4 p1 = *(const float4*)(row + kb + 4);
    const float4 p2 = *(const float4*)(row + kb + 16);
    const float4 p3 = *(const float4*)(row + kb + 20);
    float v[16] = {p0.x,p0.y,p0.z,p0.w, p1.x,p1.y,p1.z,p1.w,
                   p2.x,p2.y,p2.z,p2.w, p3.x,p3.y,p3.z,p3.w};
    v16h f;
#pragma unroll
    for (int j = 0; j < 16; ++j) f[j] = (_Float16)(do_silu ? silu_f(v[j]) : v[j]);
    return f;
}

__global__ __launch_bounds__(256) void base_gemm_wmma_fused(
    const float* __restrict__ x, const float* __restrict__ w,
    const float* __restrict__ bias, float* __restrict__ out)
{
    const int lane = threadIdx.x & 31;
    const int wave = threadIdx.x >> 5;
    const int tile = blockIdx.x * 8 + wave;
    const int tile_m = (tile & 31) * 32;
    const int tile_n = (tile >> 5) * 16;
    const int r   = lane & 15;
    const int kb0 = (lane >> 4) * 8;

    const float* xr0 = x + (size_t)(tile_m + r) * IN_F;
    const float* xr1 = xr0 + (size_t)16 * IN_F;
    const float* wr  = w + (size_t)(tile_n + r) * IN_F;

    v8f acc0 = {}, acc1 = {};
    for (int k0 = 0; k0 < IN_F; k0 += 32) {
        const int kb = k0 + kb0;
        const v16h bf = make_frag_f32(wr, kb, false);
        const v16h a0 = make_frag_f32(xr0, kb, true);
        const v16h a1 = make_frag_f32(xr1, kb, true);
        acc0 = __builtin_amdgcn_wmma_f32_16x16x32_f16(false, a0, false, bf, (short)0, acc0, false, false);
        acc1 = __builtin_amdgcn_wmma_f32_16x16x32_f16(false, a1, false, bf, (short)0, acc1, false, false);
    }
    const int n  = lane & 15;
    const int mh = (lane >> 4) * 8;
    const float bv = bias[tile_n + n];
    float* o0 = out + (size_t)(tile_m + mh) * OUT_F + tile_n + n;
    float* o1 = o0 + (size_t)16 * OUT_F;
#pragma unroll
    for (int rr = 0; rr < 8; ++rr) {
        o0[(size_t)rr * OUT_F] = acc0[rr] + bv;
        o1[(size_t)rr * OUT_F] = acc1[rr] + bv;
    }
}

// ---------------------------------------------------------------------------
// Wavelet branch, accumulated into out. 64x64 (b,o) tile per 256-thread
// block; I staged via LDS in chunks of 16; 4x4 register blocking.
// exp(-u2/2) computed as exp2(u2 * -log2(e)/2) -> single v_exp_f32.
// ---------------------------------------------------------------------------
__global__ __launch_bounds__(256) void wavelet_accum(
    const float* __restrict__ x,    // [BATCH, IN_F]
    const float* __restrict__ ww,   // [OUT_F, IN_F]
    const float* __restrict__ sc,   // [OUT_F, IN_F]
    const float* __restrict__ tr,   // [OUT_F, IN_F]
    float* __restrict__ out)        // [BATCH, OUT_F] (already base + bias)
{
    __shared__ float sx [16][64];
    __shared__ float swv[16][64];
    __shared__ float ssi[16][64];
    __shared__ float str[16][64];

    const int t  = threadIdx.x;
    const int il = t & 15;
    const int rl = t >> 4;
    const int tx = t & 15;
    const int ty = t >> 4;

    const int b0 = blockIdx.x * 64;
    const int o0 = blockIdx.y * 64;

    const float kNegHalfLog2e = -0.7213475204444817f;   // -log2(e)/2

    float acc[4][4];
#pragma unroll
    for (int a = 0; a < 4; ++a)
#pragma unroll
        for (int c = 0; c < 4; ++c) acc[a][c] = 0.0f;

    for (int i0 = 0; i0 < IN_F; i0 += 16) {
        __syncthreads();
#pragma unroll
        for (int j = 0; j < 4; ++j) {
            const int bl = rl + j * 16;
            sx[il][bl] = x[(size_t)(b0 + bl) * IN_F + i0 + il];
            const size_t pidx = (size_t)(o0 + bl) * IN_F + i0 + il;
            swv[il][bl] = ww[pidx];
            ssi[il][bl] = __builtin_amdgcn_rcpf(sc[pidx]);
            str[il][bl] = tr[pidx];
        }
        __syncthreads();

#pragma unroll
        for (int kk = 0; kk < 16; ++kk) {
            float xb[4];
#pragma unroll
            for (int jb = 0; jb < 4; ++jb) xb[jb] = sx[kk][ty + jb * 16];
#pragma unroll
            for (int jo = 0; jo < 4; ++jo) {
                const int ol = tx + jo * 16;
                const float wv = swv[kk][ol];
                const float si = ssi[kk][ol];
                const float tv = str[kk][ol];
#pragma unroll
                for (int jb = 0; jb < 4; ++jb) {
                    const float u  = (xb[jb] - tv) * si;
                    const float u2 = u * u;
                    acc[jb][jo] += (u2 - 1.0f)
                                 * __builtin_amdgcn_exp2f(u2 * kNegHalfLog2e)
                                 * wv;
                }
            }
        }
    }

#pragma unroll
    for (int jb = 0; jb < 4; ++jb) {
#pragma unroll
        for (int jo = 0; jo < 4; ++jo) {
            const size_t idx = (size_t)(b0 + ty + jb * 16) * OUT_F
                             + (o0 + tx + jo * 16);
            out[idx] += acc[jb][jo];
        }
    }
}

extern "C" void kernel_launch(void* const* d_in, const int* in_sizes, int n_in,
                              void* d_out, int out_size, void* d_ws, size_t ws_size,
                              hipStream_t stream) {
    const float* x    = (const float*)d_in[0];
    const float* bw   = (const float*)d_in[1];
    const float* ww   = (const float*)d_in[2];
    const float* sc   = (const float*)d_in[3];
    const float* tr   = (const float*)d_in[4];
    const float* bias = (const float*)d_in[5];
    float* out = (float*)d_out;

    const size_t need = (size_t)BATCH * IN_F * sizeof(_Float16)
                      + (size_t)OUT_F * IN_F * sizeof(_Float16);

    if (ws_size >= need) {
        _Float16* hx = (_Float16*)d_ws;                       // [BATCH, IN_F]
        _Float16* hwp = hx + (size_t)BATCH * IN_F;            // [OUT_F, IN_F]
        silu_cvt_f16<<<(BATCH * IN_F / 8) / 256, 256, 0, stream>>>(x, hx);
        cvt_f16<<<(OUT_F * IN_F / 8) / 256, 256, 0, stream>>>(bw, hwp);
        // 16 * 32 = 512 wave-tiles / 8 waves per block = 64 blocks.
        base_gemm_wmma_h<<<64, 256, 0, stream>>>(hx, hwp, bias, out);
    } else {
        base_gemm_wmma_fused<<<128, 256, 0, stream>>>(x, bw, bias, out);
    }

    wavelet_accum<<<dim3(BATCH / 64, OUT_F / 64), 256, 0, stream>>>(
        x, ww, sc, tr, out);
}
